// SwitchMoE_67130338837016
// MI455X (gfx1250) — matrix operand order repaired and checked by reading, not compiled, not run on hardware
//
#include <hip/hip_runtime.h>
#include <hip/hip_bf16.h>
#include <math.h>

// ---------------- types ----------------
typedef __attribute__((ext_vector_type(16))) __bf16 v16bf;
typedef __attribute__((ext_vector_type(8)))  float  v8f;
typedef int v4i __attribute__((vector_size(16)));   // matches builtin's pointee type

union FragBF16 {
    v16bf v;
    uint4 q[2];
};

static __device__ __forceinline__ unsigned short f2bf(float f) {
    unsigned int u = __float_as_uint(f);
    u += 0x7FFFu + ((u >> 16) & 1u);   // round-to-nearest-even
    return (unsigned short)(u >> 16);
}

// ---- CDNA5 async global->LDS copy (no VGPR staging, tracked by ASYNCcnt) ----
#if __has_builtin(__builtin_amdgcn_global_load_async_to_lds_b128)
#define HAVE_ASYNC_COPY 1
#else
#define HAVE_ASYNC_COPY 0
#endif

#define AS1 __attribute__((address_space(1)))
#define AS3 __attribute__((address_space(3)))

static __device__ __forceinline__ AS3 v4i* to_lds(void* p) {
    return (AS3 v4i*)p;                 // explicit addrspace cast generic -> LDS
}
static __device__ __forceinline__ AS1 v4i* to_glob(const void* p) {
    return (AS1 v4i*)p;                 // explicit addrspace cast generic -> global (drops const)
}

static __device__ __forceinline__ void wait_async0() {
#if __has_builtin(__builtin_amdgcn_s_wait_asynccnt)
    __builtin_amdgcn_s_wait_asynccnt(0);
#else
    asm volatile("s_wait_asynccnt 0" ::: "memory");
#endif
}

// ---------------- problem constants ----------------
constexpr int N_TOK = 4096;
constexpr int DIMC  = 1024;
constexpr int HID   = 4096;
constexpr int NE    = 8;

// GEMM tiling (fallback path)
constexpr int BM = 128, BN = 128, BK = 32;
constexpr int LDSK = 40;   // padded row stride (elements) for BK=32
// GEMM tiling (pre-transposed bf16 path)
constexpr int BK2 = 64;
constexpr int LK2 = 72;    // padded row stride (elements) for BK=64 (144B, 16B-multiple)

// ---------------- gate: softmax + top-1 ----------------
__global__ void gate_kernel(const float* __restrict__ x,
                            const float* __restrict__ wg,
                            const float* __restrict__ bg,
                            int* __restrict__ top_idx,
                            float* __restrict__ gval) {
    __shared__ float red[256 * NE];
    const int n = blockIdx.x;
    const int t = threadIdx.x;
    float loc[NE];
#pragma unroll
    for (int e = 0; e < NE; ++e) loc[e] = 0.0f;
    for (int k = t; k < DIMC; k += 256) {
        float xv = x[(size_t)n * DIMC + k];
        const float* wr = wg + (size_t)k * NE;
#pragma unroll
        for (int e = 0; e < NE; ++e) loc[e] += xv * wr[e];
    }
#pragma unroll
    for (int e = 0; e < NE; ++e) red[t * NE + e] = loc[e];
    __syncthreads();
    for (int s = 128; s > 0; s >>= 1) {
        if (t < s) {
#pragma unroll
            for (int e = 0; e < NE; ++e) red[t * NE + e] += red[(t + s) * NE + e];
        }
        __syncthreads();
    }
    if (t == 0) {
        float l[NE];
#pragma unroll
        for (int e = 0; e < NE; ++e) l[e] = red[e] + bg[e];
        float m = l[0]; int bi = 0;
#pragma unroll
        for (int e = 1; e < NE; ++e) { if (l[e] > m) { m = l[e]; bi = e; } }
        float s = 0.0f;
#pragma unroll
        for (int e = 0; e < NE; ++e) s += __expf(l[e] - m);
        top_idx[n] = bi;
        gval[n]    = 1.0f / s;     // softmax at the argmax: exp(0)/sum
    }
}

// ---------------- routing: deterministic counts/offsets/perm/scale ----------------
__global__ void route_kernel(const int* __restrict__ top_idx,
                             const float* __restrict__ gval,
                             int* __restrict__ offsets,     // [NE+1]
                             float* __restrict__ denom_out, // [NE]
                             int* __restrict__ perm,        // [N]
                             float* __restrict__ sscale) {  // [N]
    __shared__ int   lcnt[256 * NE];
    __shared__ float lsum[256 * NE];
    __shared__ int   ltot[NE];
    __shared__ float lden[NE];
    __shared__ int   loff[NE + 1];
    const int t = threadIdx.x;
    constexpr int CH = N_TOK / 256;   // 16 tokens per thread

    int   cnt[NE]; float ds[NE];
#pragma unroll
    for (int e = 0; e < NE; ++e) { cnt[e] = 0; ds[e] = 0.0f; }
    for (int i = 0; i < CH; ++i) {
        int n = t * CH + i;
        int e = top_idx[n];
        cnt[e]++; ds[e] += gval[n];
    }
#pragma unroll
    for (int e = 0; e < NE; ++e) { lcnt[t * NE + e] = cnt[e]; lsum[t * NE + e] = ds[e]; }
    __syncthreads();
    if (t < NE) {
        const int e = t;
        int run = 0; float s = 0.0f;
        for (int i = 0; i < 256; ++i) {             // fixed order -> deterministic
            int c = lcnt[i * NE + e];
            lcnt[i * NE + e] = run;                 // per-chunk exclusive prefix
            run += c;
            s += lsum[i * NE + e];
        }
        ltot[e] = run; lden[e] = s; denom_out[e] = s;
    }
    __syncthreads();
    if (t == 0) {
        loff[0] = 0;
        for (int e = 0; e < NE; ++e) loff[e + 1] = loff[e] + ltot[e];
        for (int i = 0; i <= NE; ++i) offsets[i] = loff[i];
    }
    __syncthreads();
    int rank[NE];
#pragma unroll
    for (int e = 0; e < NE; ++e) rank[e] = lcnt[t * NE + e];
    const float cap = (float)N_TOK;   // CAPACITY_FACTOR * N
    for (int i = 0; i < CH; ++i) {
        int n = t * CH + i;
        int e = top_idx[n];
        int pos = loff[e] + rank[e]++;
        perm[pos]   = n;
        sscale[pos] = gval[n] / (lden[e] + 1e-6f) * cap;
    }
}

// ---------------- gather x rows (fp32 -> bf16) in expert order ----------------
__global__ void gather_kernel(const float* __restrict__ x,
                              const int* __restrict__ perm,
                              unsigned short* __restrict__ xg) {
    const int i = blockIdx.x;
    const int tok = perm[i];
    const int k = threadIdx.x * 4;
    float4 f = *reinterpret_cast<const float4*>(x + (size_t)tok * DIMC + k);
    unsigned int p0 = (unsigned int)f2bf(f.x) | ((unsigned int)f2bf(f.y) << 16);
    unsigned int p1 = (unsigned int)f2bf(f.z) | ((unsigned int)f2bf(f.w) << 16);
    *reinterpret_cast<uint2*>(xg + (size_t)i * DIMC + k) = make_uint2(p0, p1);
}

// ---------------- weight pre-transpose + convert: W[e][K][NN] f32 -> Wt[e][NN][K] bf16 ----
__global__ void wt_kernel(const float* __restrict__ W,
                          unsigned short* __restrict__ Wt,
                          int K, int NN) {
    __shared__ __align__(16) unsigned short T[64 * 72];
    const int e  = blockIdx.z;
    const int n0 = blockIdx.x * 64;
    const int k0 = blockIdx.y * 64;
    const float* Wb = W + (size_t)e * K * NN;
    unsigned short* Ob = Wt + (size_t)e * NN * K;
    const int t = threadIdx.x;
#pragma unroll
    for (int it = 0; it < 4; ++it) {
        int c  = t + it * 256;
        int r  = c >> 4;            // k row within tile
        int cn = (c & 15) * 4;      // n within tile
        float4 f = *reinterpret_cast<const float4*>(Wb + (size_t)(k0 + r) * NN + n0 + cn);
        T[(cn + 0) * 72 + r] = f2bf(f.x);
        T[(cn + 1) * 72 + r] = f2bf(f.y);
        T[(cn + 2) * 72 + r] = f2bf(f.z);
        T[(cn + 3) * 72 + r] = f2bf(f.w);
    }
    __syncthreads();
    {
        int rn = t >> 2;
        int kc = (t & 3) * 16;
        uint4 a = *reinterpret_cast<const uint4*>(&T[rn * 72 + kc]);
        uint4 b = *reinterpret_cast<const uint4*>(&T[rn * 72 + kc + 8]);
        unsigned short* dst = Ob + (size_t)(n0 + rn) * K + k0 + kc;
        *reinterpret_cast<uint4*>(dst)     = a;
        *reinterpret_cast<uint4*>(dst + 8) = b;
    }
}

// ---------------- optimized GEMM: both operands bf16, K-contiguous ----------------
// MODE 0: h = gelu(Ag @ W1t[e]^T + b1[e])  -> bf16 Out
// MODE 1: out[perm[row]] = (Ag @ W2t[e]^T + b2[e]) * sscale[row] -> fp32 Out
template <int MODE>
__global__ void __launch_bounds__(256, 1)
moe_gemm_bt(const unsigned short* __restrict__ Ag,  // bf16 [N][K]
            const unsigned short* __restrict__ Bt,  // bf16 [E][NN][K]
            const float* __restrict__ bias,         // fp32 [E][NN]
            void* __restrict__ Out,
            const int* __restrict__ offsets,        // [NE+1]
            const int* __restrict__ perm,
            const float* __restrict__ sscale,
            int K, int NN) {
    const int e     = blockIdx.z;
    const int off   = offsets[e];
    const int cnt   = offsets[e + 1] - off;
    const int mbase = blockIdx.y * BM;
    if (mbase >= cnt) return;                 // uniform: EXEC stays all-ones
    const int nbase = blockIdx.x * BN;
    const int Ntot  = offsets[NE];

    __shared__ __align__(16) unsigned short As[2][BM * LK2];  // [m][k], double-buffered
    __shared__ __align__(16) unsigned short Bs[2][BN * LK2];  // [n][k], double-buffered

    const int tid   = threadIdx.x;
    const int lane  = tid & 31;
    const int wave  = tid >> 5;
    const int wm    = wave >> 2;        // 0..1
    const int wn    = wave & 3;         // 0..3
    const int lhalf = lane >> 4;        // 0/1
    const int l15   = lane & 15;

    v8f acc[4][2];
#pragma unroll
    for (int f = 0; f < 4; ++f)
#pragma unroll
        for (int g = 0; g < 2; ++g) acc[f][g] = (v8f)(0.0f);

    const unsigned short* Bb = Bt + (size_t)e * NN * K;

    auto compute = [&](int buf) {
#pragma unroll
        for (int kc = 0; kc < 2; ++kc) {       // two 32-wide K chunks
            FragBF16 bfr[2];
#pragma unroll
            for (int g = 0; g < 2; ++g) {
                int col = wn * 32 + g * 16 + l15;
                int kk  = kc * 32 + lhalf * 16;
                bfr[g].q[0] = *reinterpret_cast<const uint4*>(&Bs[buf][col * LK2 + kk]);
                bfr[g].q[1] = *reinterpret_cast<const uint4*>(&Bs[buf][col * LK2 + kk + 8]);
            }
#pragma unroll
            for (int f = 0; f < 4; ++f) {
                FragBF16 af;
                int row = wm * 64 + f * 16 + l15;
                int kk  = kc * 32 + lhalf * 16;
                af.q[0] = *reinterpret_cast<const uint4*>(&As[buf][row * LK2 + kk]);
                af.q[1] = *reinterpret_cast<const uint4*>(&As[buf][row * LK2 + kk + 8]);
#pragma unroll
                for (int g = 0; g < 2; ++g) {
                    acc[f][g] = __builtin_amdgcn_wmma_f32_16x16x32_bf16(
                        false, af.v, false, bfr[g].v, (short)0, acc[f][g], false, false);
                }
            }
        }
    };

    const int nk = K / BK2;

#if HAVE_ASYNC_COPY
    // ---- async DMA pipeline: global->LDS without VGPR staging, 1 barrier/iter ----
    auto asyncTile = [&](int k0, int buf) {
#pragma unroll
        for (int it = 0; it < 4; ++it) {
            int c = tid + it * 256;            // 1024 chunks of 8 bf16 per operand
            int row = c >> 3, kp = (c & 7) * 8;
            int grow = off + mbase + row;
            if (grow > Ntot - 1) grow = Ntot - 1;
            __builtin_amdgcn_global_load_async_to_lds_b128(
                to_glob(Ag + (size_t)grow * K + k0 + kp),
                to_lds(&As[buf][row * LK2 + kp]), 0, 0);
            __builtin_amdgcn_global_load_async_to_lds_b128(
                to_glob(Bb + (size_t)(nbase + row) * K + k0 + kp),
                to_lds(&Bs[buf][row * LK2 + kp]), 0, 0);
        }
    };
    asyncTile(0, 0);
    wait_async0();
    __syncthreads();
    for (int i = 0; i < nk; ++i) {
        const int buf = i & 1;
        if (i + 1 < nk) asyncTile((i + 1) * BK2, buf ^ 1);  // DMA overlaps compute
        compute(buf);
        if (i + 1 < nk) {
            wait_async0();       // this wave's DMA into buf^1 landed
            __syncthreads();     // everyone's landed; everyone done reading buf
        }
    }
#else
    // ---- register-staging pipeline (full register budget via launch_bounds) ----
    uint4 ar[4], br[4];
    auto loadTiles = [&](int k0) {
#pragma unroll
        for (int it = 0; it < 4; ++it) {
            int c = tid + it * 256;
            int row = c >> 3, kp = (c & 7) * 8;
            int grow = off + mbase + row;
            if (grow > Ntot - 1) grow = Ntot - 1;
            ar[it] = *reinterpret_cast<const uint4*>(Ag + (size_t)grow * K + k0 + kp);
            br[it] = *reinterpret_cast<const uint4*>(Bb + (size_t)(nbase + row) * K + k0 + kp);
        }
    };
    auto storeTiles = [&](int buf) {
#pragma unroll
        for (int it = 0; it < 4; ++it) {
            int c = tid + it * 256;
            int row = c >> 3, kp = (c & 7) * 8;
            *reinterpret_cast<uint4*>(&As[buf][row * LK2 + kp]) = ar[it];
            *reinterpret_cast<uint4*>(&Bs[buf][row * LK2 + kp]) = br[it];
        }
    };
    loadTiles(0);
    storeTiles(0);
    __syncthreads();
    for (int i = 0; i < nk; ++i) {
        const int buf = i & 1;
        if (i + 1 < nk) loadTiles((i + 1) * BK2);
        compute(buf);
        if (i + 1 < nk) {
            __syncthreads();
            storeTiles(buf ^ 1);
            __syncthreads();
        }
    }
#endif

    // --- epilogue ---
    const float* be = bias + (size_t)e * NN;
#pragma unroll
    for (int f = 0; f < 4; ++f) {
#pragma unroll
        for (int g = 0; g < 2; ++g) {
            int Nl = nbase + wn * 32 + g * 16 + l15;
#pragma unroll
            for (int r = 0; r < 8; ++r) {
                int Ml = wm * 64 + f * 16 + r + lhalf * 8;
                if (mbase + Ml < cnt) {
                    int grow = off + mbase + Ml;
                    float v = acc[f][g][r];
                    if (MODE == 0) {
                        v += be[Nl];
                        v = 0.5f * v * (1.0f + erff(v * 0.70710678118654752f));
                        ((unsigned short*)Out)[(size_t)grow * NN + Nl] = f2bf(v);
                    } else {
                        v = (v + be[Nl]) * sscale[grow];
                        int tok = perm[grow];
                        ((float*)Out)[(size_t)tok * NN + Nl] = v;
                    }
                }
            }
        }
    }
}

// ---------------- fallback GEMM (fp32 weights, fused convert; small workspace) -------
template <int MODE>
__global__ void __launch_bounds__(256, 1)
moe_gemm(const unsigned short* __restrict__ Ag,
         const float* __restrict__ Wt,
         const float* __restrict__ bias,
         void* __restrict__ Out,
         const int* __restrict__ offsets,
         const int* __restrict__ perm,
         const float* __restrict__ sscale,
         int K, int NN) {
    const int e    = blockIdx.z;
    const int off  = offsets[e];
    const int cnt  = offsets[e + 1] - off;
    const int mbase = blockIdx.y * BM;
    if (mbase >= cnt) return;
    const int nbase = blockIdx.x * BN;
    const int Ntot  = offsets[NE];

    __shared__ __align__(16) unsigned short As[BM * LDSK];
    __shared__ __align__(16) unsigned short Bs[BN * LDSK];

    const int tid   = threadIdx.x;
    const int lane  = tid & 31;
    const int wave  = tid >> 5;
    const int wm    = wave >> 2;
    const int wn    = wave & 3;
    const int lhalf = lane >> 4;
    const int l15   = lane & 15;

    v8f acc[4][2];
#pragma unroll
    for (int f = 0; f < 4; ++f)
#pragma unroll
        for (int g = 0; g < 2; ++g) acc[f][g] = (v8f)(0.0f);

    const float* Wb = Wt + (size_t)e * K * NN;

    for (int k0 = 0; k0 < K; k0 += BK) {
#pragma unroll
        for (int it = 0; it < 2; ++it) {
            int c = tid + it * 256;
            int row = c >> 2, cp = c & 3;
            int grow = off + mbase + row;
            if (grow > Ntot - 1) grow = Ntot - 1;
            uint4 d = *reinterpret_cast<const uint4*>(Ag + (size_t)grow * K + k0 + cp * 8);
            *reinterpret_cast<uint4*>(&As[row * LDSK + cp * 8]) = d;
        }
#pragma unroll
        for (int it = 0; it < 4; ++it) {
            int c = tid + it * 256;
            int kr = c >> 5;
            int np = (c & 31) * 4;
            float4 f4 = *reinterpret_cast<const float4*>(Wb + (size_t)(k0 + kr) * NN + nbase + np);
            Bs[(np + 0) * LDSK + kr] = f2bf(f4.x);
            Bs[(np + 1) * LDSK + kr] = f2bf(f4.y);
            Bs[(np + 2) * LDSK + kr] = f2bf(f4.z);
            Bs[(np + 3) * LDSK + kr] = f2bf(f4.w);
        }
        __syncthreads();

        FragBF16 bf[2];
#pragma unroll
        for (int g = 0; g < 2; ++g) {
            int col = wn * 32 + g * 16 + l15;
            int kk  = lhalf * 16;
            bf[g].q[0] = *reinterpret_cast<const uint4*>(&Bs[col * LDSK + kk]);
            bf[g].q[1] = *reinterpret_cast<const uint4*>(&Bs[col * LDSK + kk + 8]);
        }
#pragma unroll
        for (int f = 0; f < 4; ++f) {
            FragBF16 af;
            int row = wm * 64 + f * 16 + l15;
            int kk  = lhalf * 16;
            af.q[0] = *reinterpret_cast<const uint4*>(&As[row * LDSK + kk]);
            af.q[1] = *reinterpret_cast<const uint4*>(&As[row * LDSK + kk + 8]);
#pragma unroll
            for (int g = 0; g < 2; ++g) {
                acc[f][g] = __builtin_amdgcn_wmma_f32_16x16x32_bf16(
                    false, af.v, false, bf[g].v, (short)0, acc[f][g], false, false);
            }
        }
        __syncthreads();
    }

    const float* be = bias + (size_t)e * NN;
#pragma unroll
    for (int f = 0; f < 4; ++f) {
#pragma unroll
        for (int g = 0; g < 2; ++g) {
            int Nl = nbase + wn * 32 + g * 16 + l15;
#pragma unroll
            for (int r = 0; r < 8; ++r) {
                int Ml = wm * 64 + f * 16 + r + lhalf * 8;
                if (mbase + Ml < cnt) {
                    int grow = off + mbase + Ml;
                    float v = acc[f][g][r];
                    if (MODE == 0) {
                        v += be[Nl];
                        v = 0.5f * v * (1.0f + erff(v * 0.70710678118654752f));
                        ((unsigned short*)Out)[(size_t)grow * NN + Nl] = f2bf(v);
                    } else {
                        v = (v + be[Nl]) * sscale[grow];
                        int tok = perm[grow];
                        ((float*)Out)[(size_t)tok * NN + Nl] = v;
                    }
                }
            }
        }
    }
}

// ---------------- launch ----------------
extern "C" void kernel_launch(void* const* d_in, const int* in_sizes, int n_in,
                              void* d_out, int out_size, void* d_ws, size_t ws_size,
                              hipStream_t stream) {
    (void)in_sizes; (void)n_in; (void)out_size;
    const float* x  = (const float*)d_in[0];
    const float* wg = (const float*)d_in[1];
    const float* bg = (const float*)d_in[2];
    const float* W1 = (const float*)d_in[3];
    const float* b1 = (const float*)d_in[4];
    const float* W2 = (const float*)d_in[5];
    const float* b2 = (const float*)d_in[6];

    char* w = (char*)d_ws;
    float*          ws_denom  = (float*)(w + 0);          // 8 f32
    int*            ws_off    = (int*)  (w + 128);        // 9 i32
    int*            ws_top    = (int*)  (w + 256);        // 4096 i32
    float*          ws_gval   = (float*)(w + 16640);      // 4096 f32
    float*          ws_sscale = (float*)(w + 33024);      // 4096 f32
    int*            ws_perm   = (int*)  (w + 49408);      // 4096 i32
    unsigned short* ws_xg     = (unsigned short*)(w + 65792);     // 8 MB
    unsigned short* ws_hg     = (unsigned short*)(w + 8454400);   // 32 MB
    unsigned short* ws_w1t    = (unsigned short*)(w + 42008832);  // 64 MB
    unsigned short* ws_w2t    = (unsigned short*)(w + 109117696); // 64 MB
    const bool big = ws_size >= (size_t)176226560;                // end of ws_w2t

    gate_kernel  <<<N_TOK, 256, 0, stream>>>(x, wg, bg, ws_top, ws_gval);
    route_kernel <<<1,     256, 0, stream>>>(ws_top, ws_gval, ws_off, ws_denom, ws_perm, ws_sscale);
    gather_kernel<<<N_TOK, 256, 0, stream>>>(x, ws_perm, ws_xg);

    if (big) {
        wt_kernel<<<dim3(HID / 64, DIMC / 64, NE), 256, 0, stream>>>(W1, ws_w1t, DIMC, HID);
        wt_kernel<<<dim3(DIMC / 64, HID / 64, NE), 256, 0, stream>>>(W2, ws_w2t, HID, DIMC);
        moe_gemm_bt<0><<<dim3(HID / BN, N_TOK / BM, NE), 256, 0, stream>>>(
            ws_xg, ws_w1t, b1, (void*)ws_hg, ws_off, nullptr, nullptr, DIMC, HID);
        moe_gemm_bt<1><<<dim3(DIMC / BN, N_TOK / BM, NE), 256, 0, stream>>>(
            ws_hg, ws_w2t, b2, d_out, ws_off, ws_perm, ws_sscale, HID, DIMC);
    } else {
        moe_gemm<0><<<dim3(HID / BN, N_TOK / BM, NE), 256, 0, stream>>>(
            ws_xg, W1, b1, (void*)ws_hg, ws_off, nullptr, nullptr, DIMC, HID);
        moe_gemm<1><<<dim3(DIMC / BN, N_TOK / BM, NE), 256, 0, stream>>>(
            ws_hg, W2, b2, d_out, ws_off, ws_perm, ws_sscale, HID, DIMC);
    }
}
